// MemoryAttentionModule_71631464563022
// MI455X (gfx1250) — matrix operand order repaired and checked by reading, not compiled, not run on hardware
//
#include <hip/hip_runtime.h>
#include <hip/hip_bf16.h>
#include <math.h>
#include <stdint.h>

// B=512, N_SLOTS=128, IN=2048, MEM=64, OUT=2048
#define BATCH 512
#define NSLOT 128
#define IND 2048
#define MEMD 64
#define OUTD 2048

typedef __attribute__((ext_vector_type(16))) __bf16 v16bf;
typedef __attribute__((ext_vector_type(8)))  float  v8f;

// Load a 16x32 bf16 A-tile fragment for one lane (contiguous K order):
// lane<16 holds K {0..15} of row (lane&15); lane>=16 holds K {16..31}.
__device__ __forceinline__ v16bf load_a_bf16(const float* __restrict__ rp, int hi) {
    const float* p = rp + hi * 16;
    float4 a = *(const float4*)(p);
    float4 b = *(const float4*)(p + 4);
    float4 c = *(const float4*)(p + 8);
    float4 d = *(const float4*)(p + 12);
    v16bf r;
    r[0]=(__bf16)a.x; r[1]=(__bf16)a.y; r[2]=(__bf16)a.z; r[3]=(__bf16)a.w;
    r[4]=(__bf16)b.x; r[5]=(__bf16)b.y; r[6]=(__bf16)b.z; r[7]=(__bf16)b.w;
    r[8]=(__bf16)c.x; r[9]=(__bf16)c.y; r[10]=(__bf16)c.z; r[11]=(__bf16)c.w;
    r[12]=(__bf16)d.x; r[13]=(__bf16)d.y; r[14]=(__bf16)d.z; r[15]=(__bf16)d.w;
    return r;
}

// Generic WMMA GEMM: C[M,N] = concat_K(A0|A1|A2) @ B + bias, f32 in/out, bf16 MACs.
// Block = 128 threads (4 waves); wave owns a 16x16 tile; block = 64 rows x 16 cols.
// grid = (M/64, ceil(N/16)). K0,K1,K2 multiples of 32.
// VECB: B rows 16B-alignable (ldb%4==0) -> float4 staging loads.
// B loads intentionally unguarded in N (padded cols stay inside the allocation);
// stores/bias remain guarded.
template <bool VECB>
__global__ void mam_wmma_gemm(const float* __restrict__ A0, int K0, int lda0,
                              const float* __restrict__ A1, int K1, int lda1,
                              const float* __restrict__ A2, int K2, int lda2,
                              const float* __restrict__ B, int ldb,
                              const float* __restrict__ bias,
                              float* __restrict__ C, int ldc, int N) {
    __shared__ __bf16 ldsB[16 * 32];   // [n][k] col(n)-major: lane reads 16 contig K
    const int tid   = threadIdx.x;
    const int wave  = tid >> 5;
    const int lane  = tid & 31;
    const int lhalf = lane >> 4;
    const int ln    = lane & 15;
    const int colBase = blockIdx.y * 16;
    const int rowA    = blockIdx.x * 64 + wave * 16 + ln;

    // Per-thread B staging base (hoisted; no per-iteration 64-bit math).
    const float* Bp;
    int sk, sn;
    if (VECB) { sk = tid >> 2; sn = (tid & 3) * 4;          // one float4: row sk, cols sn..sn+3
                Bp = B + (size_t)sk * ldb + colBase + sn; }
    else      { sk = tid >> 4; sn = tid & 15;               // 4 floats: rows sk+{0,8,16,24}, col sn
                Bp = B + (size_t)sk * ldb + colBase + sn; }
    const size_t ldb8 = (size_t)ldb * 8;

    v8f acc = {0.f,0.f,0.f,0.f,0.f,0.f,0.f,0.f};

    auto kblock = [&](const float* __restrict__ src, int lda, int kloc, int kg) {
        __syncthreads();                       // protect previous-iter LDS reads
        const float* p = Bp + (size_t)kg * ldb;
        if (VECB) {
            float4 v = *(const float4*)p;
            ldsB[(sn + 0) * 32 + sk] = (__bf16)v.x;
            ldsB[(sn + 1) * 32 + sk] = (__bf16)v.y;
            ldsB[(sn + 2) * 32 + sk] = (__bf16)v.z;
            ldsB[(sn + 3) * 32 + sk] = (__bf16)v.w;
        } else {
            float v0 = p[0], v1 = p[ldb8], v2 = p[2 * ldb8], v3 = p[3 * ldb8];
            ldsB[sn * 32 + sk     ] = (__bf16)v0;
            ldsB[sn * 32 + sk + 8 ] = (__bf16)v1;
            ldsB[sn * 32 + sk + 16] = (__bf16)v2;
            ldsB[sn * 32 + sk + 24] = (__bf16)v3;
        }
        v16bf a = load_a_bf16(src + (size_t)rowA * lda + kloc, lhalf);
        __syncthreads();
        v16bf b = *(const v16bf*)(&ldsB[ln * 32 + lhalf * 16]);
        acc = __builtin_amdgcn_wmma_f32_16x16x32_bf16(false, a, false, b,
                                                      (short)0, acc, false, false);
    };

    for (int kg = 0; kg < K0; kg += 32) kblock(A0, lda0, kg, kg);
    for (int kg = 0; kg < K1; kg += 32) kblock(A1, lda1, kg, K0 + kg);
    for (int kg = 0; kg < K2; kg += 32) kblock(A2, lda2, kg, K0 + K1 + kg);

    const int col = colBase + ln;
    if (col < N) {
        const float bv = bias[col];
        const int rowb = blockIdx.x * 64 + wave * 16 + 8 * lhalf;
        #pragma unroll
        for (int j = 0; j < 8; ++j)
            C[(size_t)(rowb + j) * ldc + col] = acc[j] + bv;
    }
}

__device__ __forceinline__ float block_softmax_weight(float val, float* red, int tid) {
    red[tid] = val; __syncthreads();
    #pragma unroll
    for (int off = 64; off > 0; off >>= 1) {
        if (tid < off) red[tid] = fmaxf(red[tid], red[tid + off]);
        __syncthreads();
    }
    float mx = red[0]; __syncthreads();
    float e = __expf(val - mx);
    red[tid] = e; __syncthreads();
    #pragma unroll
    for (int off = 64; off > 0; off >>= 1) {
        if (tid < off) red[tid] += red[tid + off];
        __syncthreads();
    }
    float s = red[0]; __syncthreads();
    return e / s;
}

// One block per batch row: scores, two softmaxes over 128 slots, weighted sums.
// memory[b] (32KB) staged into LDS via CDNA5 async global->LDS DMA (ASYNCcnt),
// bypassing the VGPR file; per-lane LDS addresses implement the stride-65 pad.
__global__ void mam_attention(const float* __restrict__ memory,
                              const float* __restrict__ q,
                              const float* __restrict__ W_pr,
                              const float* __restrict__ b_pr,
                              float* __restrict__ active,
                              float* __restrict__ passive) {
    __shared__ float mem[NSLOT * 65];   // stride 65 -> no LDS bank conflicts
    __shared__ float qv[MEMD], wpr[MEMD];
    __shared__ float wsc[NSLOT], wpc[NSLOT], red[NSLOT];
    const int tid = threadIdx.x;
    const int b   = blockIdx.x;
    const float* mb = memory + (size_t)b * NSLOT * MEMD;

    #pragma unroll
    for (int i = 0; i < 16; ++i) {            // 128 thr * 16 * 16B = 32KB
        int e  = tid + i * 128;               // float4 index within memory[b]
        int fi = e * 4;
        int n = fi >> 6, m = fi & 63;         // 64%4==0: b128 never crosses a row
        uint32_t ldsa = (uint32_t)(uintptr_t)(&mem[n * 65 + m]);  // LDS byte offset
        uint32_t voff = (uint32_t)fi * 4u;    // byte offset from saddr base
        asm volatile("global_load_async_to_lds_b128 %0, %1, %2"
                     :: "v"(ldsa), "v"(voff), "s"(mb)
                     : "memory");
    }
    if (tid < MEMD) qv[tid] = q[b * MEMD + tid];
    else            wpr[tid - MEMD] = W_pr[tid - MEMD];
    asm volatile("s_wait_asynccnt 0x0" ::: "memory");
    __syncthreads();

    float s = 0.f, t = 0.f;
    const float* mr = &mem[tid * 65];
    #pragma unroll 8
    for (int m = 0; m < MEMD; ++m) { float v = mr[m]; s += v * qv[m]; t += v * wpr[m]; }
    t += b_pr[0];

    float w = block_softmax_weight(s, red, tid);   // active-recall weights
    float p = block_softmax_weight(t, red, tid);   // passive-recall weights
    wsc[tid] = w; wpc[tid] = p;
    __syncthreads();

    if (tid < MEMD) {
        float ar = 0.f, pr = 0.f;
        #pragma unroll 8
        for (int n = 0; n < NSLOT; ++n) {
            float v = mem[n * 65 + tid];
            ar += wsc[n] * v; pr += wpc[n] * v;
        }
        active[b * MEMD + tid]  = ar;
        passive[b * MEMD + tid] = pr;
    }
}

// Gated memory update: rem = xr[b] + mem@W_r[2048:] ; new = mem*forget + rem[64]*rem[:64]
// Each wave: one 16-row tile of (B*N)=65536 rows; 5 col-tiles (cols 0..79, 65 valid).
__global__ void mam_memupdate(const float* __restrict__ memory,
                              const float* __restrict__ W_f,
                              const float* __restrict__ W_r,
                              const float* __restrict__ xf,
                              const float* __restrict__ xr,
                              float* __restrict__ newmem) {
    __shared__ __bf16 ldsB[80 * 64];   // W_r[2048:2112, 0:65] col-major, padded to 80 cols
    __shared__ float  wf[MEMD];
    const int tid = threadIdx.x;

    // contiguous read of the 64x65 memory-part slab of W_r, transpose into LDS
    const float* wrm = W_r + (size_t)IND * (MEMD + 1);
    for (int j = tid; j < MEMD * (MEMD + 1); j += 128) {
        int k = j / (MEMD + 1), col = j - k * (MEMD + 1);
        ldsB[col * 64 + k] = (__bf16)wrm[j];
    }
    for (int j = tid; j < 15 * 64; j += 128)      // zero pad cols 65..79
        ldsB[65 * 64 + j] = (__bf16)0.0f;
    if (tid < MEMD) wf[tid] = W_f[IND + tid];
    __syncthreads();

    const int wave  = tid >> 5;
    const int lane  = tid & 31;
    const int lhalf = lane >> 4;
    const int ln    = lane & 15;
    const int rowbase = (blockIdx.x * 4 + wave) * 16;   // flat b*128+n, tile within one b
    const int bidx    = rowbase >> 7;
    const float* rp   = memory + (size_t)(rowbase + ln) * MEMD;

    v8f acc[5] = {};
    #pragma unroll
    for (int kb = 0; kb < 2; ++kb) {
        v16bf a = load_a_bf16(rp + kb * 32, lhalf);
        #pragma unroll
        for (int t = 0; t < 5; ++t) {
            v16bf bv = *(const v16bf*)(&ldsB[(t * 16 + ln) * 64 + kb * 32 + lhalf * 16]);
            acc[t] = __builtin_amdgcn_wmma_f32_16x16x32_bf16(false, a, false, bv,
                                                             (short)0, acc[t], false, false);
        }
    }
    // add x-side partial (already contains b_r); uniform over rows of the tile
    #pragma unroll
    for (int t = 0; t < 5; ++t) {
        int col = t * 16 + ln;
        float xv = (col < MEMD + 1) ? xr[bidx * (MEMD + 1) + col] : 0.0f;
        #pragma unroll
        for (int j = 0; j < 8; ++j) acc[t][j] += xv;
    }
    // forget gate for row (rowbase + ln): xf includes b_f and x-part dot
    float z = xf[bidx];
    #pragma unroll 8
    for (int m = 0; m < MEMD; ++m) z += rp[m] * wf[m];
    float f = 1.1f / (1.0f + __expf(-z));

    // gate scalar rem[row,64] sits in acc[4] col 0 -> lanes 0 (rows j) / 16 (rows j+8)
    #pragma unroll
    for (int t = 0; t < 4; ++t) {
        int col = t * 16 + ln;
        #pragma unroll
        for (int j = 0; j < 8; ++j) {
            int   rl = j + 8 * lhalf;
            float g  = __shfl(acc[4][j], lhalf * 16, 32);
            float fj = __shfl(f, rl, 32);
            size_t idx = (size_t)(rowbase + rl) * MEMD + col;
            newmem[idx] = memory[idx] * fj + g * acc[t][j];
        }
    }
}

extern "C" void kernel_launch(void* const* d_in, const int* in_sizes, int n_in,
                              void* d_out, int out_size, void* d_ws, size_t ws_size,
                              hipStream_t stream) {
    (void)in_sizes; (void)n_in; (void)out_size; (void)ws_size;
    const float* x    = (const float*)d_in[0];
    const float* mem  = (const float*)d_in[1];
    const float* W_ar = (const float*)d_in[2];
    const float* b_ar = (const float*)d_in[3];
    const float* W_pr = (const float*)d_in[4];
    const float* b_pr = (const float*)d_in[5];
    const float* W_f  = (const float*)d_in[6];
    const float* b_f  = (const float*)d_in[7];
    const float* W_r  = (const float*)d_in[8];
    const float* b_r  = (const float*)d_in[9];
    const float* W_o  = (const float*)d_in[10];
    const float* b_o  = (const float*)d_in[11];

    float* out    = (float*)d_out;               // [512, 2048]
    float* newmem = out + (size_t)BATCH * OUTD;  // [512, 128, 64]

    float* ws      = (float*)d_ws;
    float* q       = ws;                          // [512, 64]
    float* xr      = q  + BATCH * MEMD;           // [512, 65]
    float* xf      = xr + BATCH * (MEMD + 1);     // [512]
    float* active  = xf + BATCH;                  // [512, 64]
    float* passive = active + BATCH * MEMD;       // [512, 64]

    // 1) x-side projections (bias folded in)
    mam_wmma_gemm<true><<<dim3(BATCH/64, MEMD/16), 128, 0, stream>>>(
        x, IND, IND, nullptr, 0, 0, nullptr, 0, 0, W_ar, MEMD, b_ar, q, MEMD, MEMD);
    mam_wmma_gemm<false><<<dim3(BATCH/64, 5), 128, 0, stream>>>(
        x, IND, IND, nullptr, 0, 0, nullptr, 0, 0, W_r, MEMD + 1, b_r, xr, MEMD + 1, MEMD + 1);
    mam_wmma_gemm<false><<<dim3(BATCH/64, 1), 128, 0, stream>>>(
        x, IND, IND, nullptr, 0, 0, nullptr, 0, 0, W_f, 1, b_f, xf, 1, 1);

    // 2) softmax attention over slots -> active/passive recall (async LDS staging)
    mam_attention<<<BATCH, 128, 0, stream>>>(mem, q, W_pr, b_pr, active, passive);

    // 3) output = [x | active | passive] @ W_o + b_o (virtual concat, 32-aligned splits)
    mam_wmma_gemm<true><<<dim3(BATCH/64, OUTD/16), 128, 0, stream>>>(
        x, IND, IND, active, MEMD, MEMD, passive, MEMD, MEMD,
        W_o, OUTD, b_o, out, OUTD, OUTD);

    // 4) gated memory update (factored stack@W_r / stack@W_f)
    mam_memupdate<<<(BATCH * NSLOT) / 64, 128, 0, stream>>>(
        mem, W_f, W_r, xf, xr, newmem);
}
